// SimpleGatedGCNLayer_83476984365559
// MI455X (gfx1250) — matrix operand order repaired and checked by reading, not compiled, not run on hardware
//
#include <hip/hip_runtime.h>

#define DD 128

typedef __attribute__((ext_vector_type(16))) __bf16 v16bf;
typedef __attribute__((ext_vector_type(2)))  __bf16 v2bf;
typedef __attribute__((ext_vector_type(8)))  float  v8f;

union BF16x16 { v16bf v; unsigned u[8]; };

// pack two f32 into a packed bf16 pair
#if __has_builtin(__builtin_amdgcn_cvt_pk_bf16_f32)
__device__ __forceinline__ unsigned pk2(float a, float b) {
  v2bf r = __builtin_amdgcn_cvt_pk_bf16_f32(a, b);
  return __builtin_bit_cast(unsigned, r);
}
#else
__device__ __forceinline__ unsigned pk2(float a, float b) {
  // truncating pack (RTZ): 3 VALU ops, accuracy noise below bf16 rounding level
  return (__float_as_uint(b) & 0xFFFF0000u) | (__float_as_uint(a) >> 16);
}
#endif

__device__ __forceinline__ unsigned short f2bf(float a) {
  unsigned ua = __float_as_uint(a);
  ua = (ua + 0x7FFFu + ((ua >> 16) & 1u)) >> 16;
  return (unsigned short)ua;
}

// load one 16x16x32 B operand (column n, K-chunk) from padded LDS
__device__ __forceinline__ void loadB(const unsigned short* wrow, int k0, BF16x16& B) {
  uint4 q0 = *(const uint4*)(wrow + k0);
  uint4 q1 = *(const uint4*)(wrow + k0 + 16);
  B.u[0] = q0.x; B.u[1] = q0.y; B.u[2] = q0.z; B.u[3] = q0.w;
  B.u[4] = q1.x; B.u[5] = q1.y; B.u[6] = q1.z; B.u[7] = q1.w;
}

// ---------------- prep: transpose 5 weights to bf16, layout Wt[m][n][k] ----------------
__global__ void gcn_prep_weights(const float* __restrict__ Wa, const float* __restrict__ Wb,
                                 const float* __restrict__ Wc, const float* __restrict__ Wd,
                                 const float* __restrict__ We, unsigned short* __restrict__ Wt) {
  int tid = blockIdx.x * blockDim.x + threadIdx.x;
  if (tid >= 5 * DD * DD) return;
  int m = tid / (DD * DD);
  int rem = tid - m * DD * DD;
  int n = rem >> 7;
  int k = rem & 127;
  const float* W = (m == 0) ? Wa : (m == 1) ? Wb : (m == 2) ? Wc : (m == 3) ? Wd : We;
  Wt[tid] = f2bf(W[k * DD + n]);
}

// ---------------- zero num/den accumulators ----------------
__global__ void gcn_zero(float4* __restrict__ p, int n4) {
  int i = blockIdx.x * blockDim.x + threadIdx.x;
  if (i < n4) p[i] = make_float4(0.f, 0.f, 0.f, 0.f);
}

// ---------------- node GEMMs: Ah,Bh,Dh,Eh = (h*norm)@W + b ----------------
__global__ __launch_bounds__(256) void gcn_node_gemm(
    const float* __restrict__ h, const float* __restrict__ norm,
    const unsigned short* __restrict__ Wt,
    const float* __restrict__ ba, const float* __restrict__ bb,
    const float* __restrict__ bd, const float* __restrict__ be,
    float* __restrict__ Ah, float* __restrict__ Bh,
    float* __restrict__ Dh, float* __restrict__ Eh,
    float* __restrict__ dump, int N) {
  __shared__ __align__(16) unsigned short wlds[DD * 136];  // padded rows: bank spread
  const int t    = threadIdx.x;
  const int wave = t >> 5, lane = t & 31;
  const int r    = lane & 15;
  const int kbA  = (lane >> 4) << 3;   // A layout: half-wave K offset 0/8
  const int kbB  = (lane >> 4) << 3;   // B layout: half-wave K offset 0/8
  const int ln   = lane & 15;
  const int mo   = (lane >> 4) << 3;   // C/D row offset 0/8

  const int rowBase = blockIdx.x * 128 + wave * 16;
  int row = rowBase + r; if (row >= N) row = N - 1;
  const float nrm = norm[row];
  const float* hrow = h + (size_t)row * DD;

  // build bf16 A operands for K chunks 0,32,64,96 (hs = h*norm, converted)
  BF16x16 A[4];
#pragma unroll
  for (int c = 0; c < 4; ++c) {
    int k0 = c * 32 + kbA;
    float4 f0 = *(const float4*)(hrow + k0);
    float4 f1 = *(const float4*)(hrow + k0 + 4);
    float4 f2 = *(const float4*)(hrow + k0 + 16);
    float4 f3 = *(const float4*)(hrow + k0 + 20);
    A[c].u[0] = pk2(f0.x * nrm, f0.y * nrm);
    A[c].u[1] = pk2(f0.z * nrm, f0.w * nrm);
    A[c].u[2] = pk2(f1.x * nrm, f1.y * nrm);
    A[c].u[3] = pk2(f1.z * nrm, f1.w * nrm);
    A[c].u[4] = pk2(f2.x * nrm, f2.y * nrm);
    A[c].u[5] = pk2(f2.z * nrm, f2.w * nrm);
    A[c].u[6] = pk2(f3.x * nrm, f3.y * nrm);
    A[c].u[7] = pk2(f3.z * nrm, f3.w * nrm);
  }

  const int   matsel[4] = {0, 1, 3, 4};          // Wa, Wb, Wd, We in Wt
  const float* biasArr[4] = {ba, bb, bd, be};
  float*       outArr[4]  = {Ah, Bh, Dh, Eh};

#pragma unroll
  for (int mm = 0; mm < 4; ++mm) {
    __syncthreads();                              // protect LDS reuse
    const unsigned short* wsrc = Wt + matsel[mm] * DD * DD;
    for (int i = t; i < DD * DD; i += 256)
      wlds[(i >> 7) * 136 + (i & 127)] = wsrc[i];
    __syncthreads();

    const float* bias = biasArr[mm];
    float*       outp = outArr[mm];
    for (int nt = 0; nt < 8; ++nt) {
      const int n = nt * 16 + ln;
      const unsigned short* wrow = &wlds[n * 136];
      // preload all 4 K-chunks of B, then issue 4 back-to-back WMMAs
      BF16x16 B[4];
#pragma unroll
      for (int kc = 0; kc < 4; ++kc) loadB(wrow, kc * 32 + kbB, B[kc]);
      v8f acc = {0.f, 0.f, 0.f, 0.f, 0.f, 0.f, 0.f, 0.f};
#pragma unroll
      for (int kc = 0; kc < 4; ++kc)
        acc = __builtin_amdgcn_wmma_f32_16x16x32_bf16(false, A[kc].v, false, B[kc].v,
                                                      (short)0, acc, false, false);
      const float bn = bias[n];
#pragma unroll
      for (int i = 0; i < 8; ++i) {   // branchless: OOB rows go to dump row
        int rr = rowBase + mo + i;
        float* p = (rr < N) ? (outp + (size_t)rr * DD + n) : (dump + n);
        *p = acc[i] + bn;
      }
    }
  }
}

// ---------------- edge kernel: Ce GEMM + gather + sigmoid + atomic segment-sum ----------------
__global__ __launch_bounds__(256) void gcn_edge(
    const float* __restrict__ e, const int* __restrict__ src, const int* __restrict__ dst,
    const unsigned short* __restrict__ Wt, const float* __restrict__ bc,
    const float* __restrict__ Bh, const float* __restrict__ Dh, const float* __restrict__ Eh,
    float* __restrict__ eout, float* __restrict__ num, float* __restrict__ den,
    float* __restrict__ dump, int E_) {
  __shared__ __align__(16) unsigned short wlds[DD * 136];
  const int t = threadIdx.x;
  const unsigned short* wsrc = Wt + 2 * DD * DD;   // Wc
  for (int i = t; i < DD * DD; i += 256)
    wlds[(i >> 7) * 136 + (i & 127)] = wsrc[i];
  __syncthreads();

  const int wave = t >> 5, lane = t & 31;
  const int r   = lane & 15;
  const int kbA = (lane >> 4) << 3;
  const int kbB = (lane >> 4) << 3;
  const int ln  = lane & 15;
  const int mo  = (lane >> 4) << 3;

  const int edgeBase = blockIdx.x * 128 + wave * 16;
  int erow = edgeBase + r; if (erow >= E_) erow = E_ - 1;
  const float* erp = e + (size_t)erow * DD;

  BF16x16 A[4];
#pragma unroll
  for (int c = 0; c < 4; ++c) {
    int k0 = c * 32 + kbA;
    float4 f0 = *(const float4*)(erp + k0);
    float4 f1 = *(const float4*)(erp + k0 + 4);
    float4 f2 = *(const float4*)(erp + k0 + 16);
    float4 f3 = *(const float4*)(erp + k0 + 20);
    A[c].u[0] = pk2(f0.x, f0.y);
    A[c].u[1] = pk2(f0.z, f0.w);
    A[c].u[2] = pk2(f1.x, f1.y);
    A[c].u[3] = pk2(f1.z, f1.w);
    A[c].u[4] = pk2(f2.x, f2.y);
    A[c].u[5] = pk2(f2.z, f2.w);
    A[c].u[6] = pk2(f3.x, f3.y);
    A[c].u[7] = pk2(f3.z, f3.w);
  }

  // hoist src/dst row offsets for this lane's 8 output rows (clamped)
  int sO[8], dO[8];
#pragma unroll
  for (int i = 0; i < 8; ++i) {
    int eidx = edgeBase + mo + i; if (eidx >= E_) eidx = E_ - 1;
    sO[i] = src[eidx] * DD;
    dO[i] = dst[eidx] * DD;
  }

  for (int nt = 0; nt < 8; ++nt) {
    const int n = nt * 16 + ln;
    const unsigned short* wrow = &wlds[n * 136];
    BF16x16 B[4];
#pragma unroll
    for (int kc = 0; kc < 4; ++kc) loadB(wrow, kc * 32 + kbB, B[kc]);
    v8f acc = {0.f, 0.f, 0.f, 0.f, 0.f, 0.f, 0.f, 0.f};
#pragma unroll
    for (int kc = 0; kc < 4; ++kc)
      acc = __builtin_amdgcn_wmma_f32_16x16x32_bf16(false, A[kc].v, false, B[kc].v,
                                                    (short)0, acc, false, false);
    const float bn = bc[n];
#pragma unroll
    for (int i = 0; i < 8; ++i) {     // branchless: OOB edges write/accumulate into dump
      int eidx = edgeBase + mo + i;
      bool ok = eidx < E_;
      float eij = acc[i] + bn + Dh[sO[i] + n] + Eh[dO[i] + n];
      float* pe = ok ? (eout + (size_t)eidx * DD + n) : (dump + n);
      *pe = eij;
      float sg = 1.0f / (1.0f + __expf(-eij));
      float* pn = ok ? (num + dO[i] + n) : (dump + n);
      float* pd = ok ? (den + dO[i] + n) : (dump + n);
      atomicAdd(pn, sg * Bh[sO[i] + n]);
      atomicAdd(pd, sg);
    }
  }
}

// ---------------- finalize: h_out = where(den>0, Ah + num/(den+1e-6), hs) * norm ----------------
__global__ void gcn_finalize(const float* __restrict__ h, const float* __restrict__ norm,
                             const float* __restrict__ Ah, const float* __restrict__ num,
                             const float* __restrict__ den, float* __restrict__ hout, int N) {
  int i = blockIdx.x * blockDim.x + threadIdx.x;
  int total = N * (DD / 4);
  if (i >= total) return;
  int row = i >> 5;
  int c = (i & 31) * 4;
  float nr = norm[row];
  size_t off = (size_t)row * DD + c;
  float4 hv = *(const float4*)(h + off);
  float4 nm = *(const float4*)(num + off);
  float4 dn = *(const float4*)(den + off);
  float4 av = *(const float4*)(Ah + off);
  float4 o;
  o.x = (dn.x > 0.f ? (av.x + nm.x / (dn.x + 1e-6f)) : hv.x * nr) * nr;
  o.y = (dn.y > 0.f ? (av.y + nm.y / (dn.y + 1e-6f)) : hv.y * nr) * nr;
  o.z = (dn.z > 0.f ? (av.z + nm.z / (dn.z + 1e-6f)) : hv.z * nr) * nr;
  o.w = (dn.w > 0.f ? (av.w + nm.w / (dn.w + 1e-6f)) : hv.w * nr) * nr;
  *(float4*)(hout + off) = o;
}

extern "C" void kernel_launch(void* const* d_in, const int* in_sizes, int n_in,
                              void* d_out, int out_size, void* d_ws, size_t ws_size,
                              hipStream_t stream) {
  (void)n_in; (void)out_size; (void)ws_size;
  const float* h    = (const float*)d_in[0];
  const float* e    = (const float*)d_in[1];
  const float* norm = (const float*)d_in[2];
  const int*   src  = (const int*)d_in[3];
  const int*   dst  = (const int*)d_in[4];
  const float* Wa = (const float*)d_in[5];  const float* ba = (const float*)d_in[6];
  const float* Wb = (const float*)d_in[7];  const float* bb = (const float*)d_in[8];
  const float* Wc = (const float*)d_in[9];  const float* bc = (const float*)d_in[10];
  const float* Wd = (const float*)d_in[11]; const float* bd = (const float*)d_in[12];
  const float* We = (const float*)d_in[13]; const float* be = (const float*)d_in[14];

  const int N = in_sizes[0] / DD;
  const int E = in_sizes[1] / DD;

  float* hout = (float*)d_out;
  float* eout = (float*)d_out + (size_t)N * DD;

  // workspace layout
  char* ws = (char*)d_ws;
  unsigned short* Wt = (unsigned short*)ws;
  size_t o = (size_t)5 * DD * DD * sizeof(unsigned short);
  o = (o + 255) & ~(size_t)255;
  const size_t matBytes = (size_t)N * DD * sizeof(float);
  float* Ah  = (float*)(ws + o); o += matBytes;
  float* Bh  = (float*)(ws + o); o += matBytes;
  float* Dh  = (float*)(ws + o); o += matBytes;
  float* Eh  = (float*)(ws + o); o += matBytes;
  float* num = (float*)(ws + o); o += matBytes;   // num, den contiguous for one zero pass
  float* den = (float*)(ws + o); o += matBytes;
  float* dump = (float*)(ws + o); o += DD * sizeof(float);  // sink for OOB lanes

  gcn_prep_weights<<<(5 * DD * DD + 255) / 256, 256, 0, stream>>>(Wa, Wb, Wc, Wd, We, Wt);

  int zn4 = (2 * N * DD) / 4;
  gcn_zero<<<(zn4 + 255) / 256, 256, 0, stream>>>((float4*)num, zn4);

  gcn_node_gemm<<<(N + 127) / 128, 256, 0, stream>>>(h, norm, Wt, ba, bb, bd, be,
                                                     Ah, Bh, Dh, Eh, dump, N);

  gcn_edge<<<(E + 127) / 128, 256, 0, stream>>>(e, src, dst, Wt, bc, Bh, Dh, Eh,
                                                eout, num, den, dump, E);

  gcn_finalize<<<(N * (DD / 4) + 255) / 256, 256, 0, stream>>>(h, norm, Ah, num, den, hout, N);
}
